// HungarianMatcher_10110353015336
// MI455X (gfx1250) — matrix operand order repaired and checked by reading, not compiled, not run on hardware
//
#include <hip/hip_runtime.h>

typedef __attribute__((ext_vector_type(16))) _Float16 v16h;
typedef __attribute__((ext_vector_type(8)))  float    v8f;

#define T_      6
#define B_      2
#define Q_      50
#define P_      100          // B*Q
#define HS      72
#define WSRC    72
#define N_      8
#define H_      288
#define W_      288
#define NTILES  7            // ceil(112/16) row tiles of P (padded to 112)
#define NPIX    (H_*W_)      // 82944
#define NCHUNK  (NPIX/32)    // 2592 chunks of 32 pixels; 288/32=9 chunks per row
#define CSPLIT  32           // pixel-split blocks per (t,tile)

// workspace layout (floats)
#define OFF_ACCL  0
#define SZ_ACC    (T_*NTILES*16*16)        // 10752: C tiles (col 8 = p.sum)
#define OFF_ACCS  (OFF_ACCL + SZ_ACC)
#define OFF_IOUN  (OFF_ACCS + SZ_ACC)
#define SZ_IOU    (T_*NTILES*16)           // 672
#define OFF_IOUD  (OFF_IOUN + SZ_IOU)
#define OFF_GSUM  (OFF_IOUD + SZ_IOU)
#define WS_FLOATS (OFF_GSUM + T_*N_)       // 22896 floats ≈ 90 KB

__device__ __forceinline__ float sigmoidf_(float v) {
    return 1.0f / (1.0f + __expf(-v));
}

__global__ __launch_bounds__(256) void zero_ws_kernel(float* ws, int n) {
    int i = blockIdx.x * 256 + threadIdx.x;
    if (i < n) ws[i] = 0.0f;
}

// g.sum per (t,n): 48 blocks, tree-reduce 82944 values each
__global__ __launch_bounds__(256) void gsum_kernel(const float* __restrict__ tgt,
                                                   float* __restrict__ ws) {
    const int tn = blockIdx.x;                      // t*8+n
    const float* g = tgt + (size_t)tn * NPIX;
    float s = 0.0f;
    for (int i = threadIdx.x; i < NPIX; i += 256) s += g[i];
    __shared__ float red[256];
    red[threadIdx.x] = s;
    __syncthreads();
    for (int o = 128; o > 0; o >>= 1) {
        if (threadIdx.x < (unsigned)o) red[threadIdx.x] += red[threadIdx.x + o];
        __syncthreads();
    }
    if (threadIdx.x == 0) ws[OFF_GSUM + tn] = red[0];
}

// Fused: bilinear 72->288 upsample + sigmoid -> f16 A-fragment, tgt -> B-fragment,
// WMMA f32_16x16x32_f16 pairwise dice accumulation (B col 8 = ones => C col 8 = p.sum),
// lane-local thresholded-IoU sums. grid = (T*NTILES, CSPLIT), 256 threads (8 waves).
__global__ __launch_bounds__(256) void fused_dice_wmma_kernel(
    const float* __restrict__ predL, const float* __restrict__ predS,
    const float* __restrict__ tgt,   float* __restrict__ ws)
{
    const int tp   = blockIdx.x;          // 0..41
    const int t    = tp / NTILES;
    const int tile = tp % NTILES;
    const int lane = threadIdx.x & 31;
    const int wv   = threadIdx.x >> 5;    // wave in WG, 0..7
    const int lhi  = lane >> 4;           // lane half
    const int m    = lane & 15;           // A row within tile / C column index too

    // A-side: this lane's query row
    const int p = tile * 16 + m;
    const bool valid = (p < P_);
    const int b = valid ? (p / Q_) : 0;
    const int q = valid ? (p % Q_) : 0;
    const size_t mbase = (((size_t)t * B_ + b) * Q_ + q) * (HS * WSRC);
    const float* mL = predL + mbase;
    const float* mS = predS + mbase;

    // B-side: this lane's target column n, K half
    const int n     = lane & 15;
    const int khalf = lhi * 16;
    const float* grow = tgt + ((size_t)t * N_ + (n < N_ ? n : 0)) * NPIX;

    v8f cL = {0.f, 0.f, 0.f, 0.f, 0.f, 0.f, 0.f, 0.f};
    v8f cS = {0.f, 0.f, 0.f, 0.f, 0.f, 0.f, 0.f, 0.f};
    float iouN = 0.0f, iouD = 0.0f;

    const int slot = blockIdx.y * 8 + wv;           // 0..255 wave slots
    for (int c = slot; c < NCHUNK; c += CSPLIT * 8) {
        const int y  = c / 9;                       // all 32 pixels share row y
        const int x0 = (c % 9) * 32;

        // vertical bilinear taps (align_corners=False), chunk-uniform
        const float fy  = (y + 0.5f) * 0.25f - 0.5f;
        const float y0f = floorf(fy);
        const float wy  = fy - y0f;
        int y0 = (int)y0f;
        int y1 = y0 + 1;
        y0 = min(max(y0, 0), HS - 1);
        y1 = min(max(y1, 0), HS - 1);
        const float* rL0 = mL + y0 * WSRC; const float* rL1 = mL + y1 * WSRC;
        const float* rS0 = mS + y0 * WSRC; const float* rS1 = mS + y1 * WSRC;

        // ---- A fragments: 16 f16 values per lane (ISA 16-bit A 16x32 layout) ----
        v16h aL, aS;
        #pragma unroll
        for (int e = 0; e < 16; ++e) {
            const int K = (e < 8) ? (lhi * 8 + e) : (16 + lhi * 8 + (e - 8));
            const int x = x0 + K;
            const float fx  = x * 0.25f - 0.375f;   // (x+0.5)/4 - 0.5
            const float x0f = floorf(fx);
            const float wx  = fx - x0f;
            int xa = (int)x0f;
            int xb = xa + 1;
            xa = min(max(xa, 0), WSRC - 1);
            xb = min(max(xb, 0), WSRC - 1);
            const float w00 = (1.f - wy) * (1.f - wx), w01 = (1.f - wy) * wx;
            const float w10 = wy * (1.f - wx),         w11 = wy * wx;
            float sL = 0.0f, sS = 0.0f;
            if (valid) {
                const float vL = w00 * rL0[xa] + w01 * rL0[xb] + w10 * rL1[xa] + w11 * rL1[xb];
                const float vS = w00 * rS0[xa] + w01 * rS0[xb] + w10 * rS1[xa] + w11 * rS1[xb];
                sL = sigmoidf_(vL);
                sS = sigmoidf_(vS);
            }
            aL[e] = (_Float16)sL;
            aS[e] = (_Float16)sS;
            // conditioned IoU (thresholded, f32, lane-local per row m)
            const float pl = (sL > 0.5f) ? sL : 0.0f;
            const float ps = (sS > 0.5f) ? sS : 0.0f;
            iouN += pl * ps;
            iouD += pl;
        }

    // ---- B fragment: K in table order (8*lhi+e, 16+8*lhi+e-8) of target column n ----
        v16h bv;
        if (n < N_) {
            const float* gp = grow + (size_t)y * W_ + x0 + lhi * 8;
            #pragma unroll
            for (int e = 0; e < 16; ++e) bv[e] = (_Float16)gp[(e < 8) ? e : (e + 8)];
        } else if (n == N_) {                        // ones column -> C[:,8] = p.sum
            #pragma unroll
            for (int e = 0; e < 16; ++e) bv[e] = (_Float16)1.0f;
        } else {
            #pragma unroll
            for (int e = 0; e < 16; ++e) bv[e] = (_Float16)0.0f;
        }

        // uniform control flow here: EXEC all-ones for WMMA
        cL = __builtin_amdgcn_wmma_f32_16x16x32_f16(false, aL, false, bv,
                                                    (short)0, cL, false, false);
        cS = __builtin_amdgcn_wmma_f32_16x16x32_f16(false, aS, false, bv,
                                                    (short)0, cS, false, false);
    }

    // combine partial C tiles across the 256 contributing waves per (t,tile)
    float* accL = ws + OFF_ACCL;
    float* accS = ws + OFF_ACCS;
    #pragma unroll
    for (int j = 0; j < 8; ++j) {
        const int row = j + 8 * lhi;                 // C layout: VGPR j, lane half
        const int idx = ((t * NTILES + tile) * 16 + row) * 16 + n;
        atomicAdd(&accL[idx], cL[j]);
        atomicAdd(&accS[idx], cS[j]);
    }
    atomicAdd(&ws[OFF_IOUN + (t * NTILES + tile) * 16 + m], iouN);
    atomicAdd(&ws[OFF_IOUD + (t * NTILES + tile) * 16 + m], iouD);
}

// final combine: dice + iou + is_referred softmax cost -> C[B,Q,N]
__global__ __launch_bounds__(256) void final_combine_kernel(
    const float* __restrict__ refL, const float* __restrict__ refS,
    const int* __restrict__ refIdx, const float* __restrict__ ws,
    float* __restrict__ out)
{
    const int tid = blockIdx.x * 256 + threadIdx.x;
    if (tid >= P_ * N_) return;
    const int p = tid / N_, n = tid % N_;
    const int b = p / Q_,  q = p % Q_;
    const int tile = p / 16, m = p % 16;

    const int rg0 = refIdx[0] + 0 * (N_ / B_);
    const int rg1 = refIdx[1] + 1 * (N_ / B_);
    const bool isref = (n == rg0) || (n == rg1);

    float costRef = 0.f, diceL = 0.f, diceS = 0.f, iou = 0.f;
    const float* accL = ws + OFF_ACCL;
    const float* accS = ws + OFF_ACCS;
    for (int t = 0; t < T_; ++t) {
        const size_t ri = (((size_t)t * B_ + b) * Q_ + q) * 2;
        {   // long head softmax over 2 classes
            const float a0 = refL[ri], a1 = refL[ri + 1];
            const float mx = fmaxf(a0, a1);
            const float e0 = __expf(a0 - mx), e1 = __expf(a1 - mx);
            costRef -= isref ? (e0 / (e0 + e1)) : (e1 / (e0 + e1));
        }
        {   // short head
            const float a0 = refS[ri], a1 = refS[ri + 1];
            const float mx = fmaxf(a0, a1);
            const float e0 = __expf(a0 - mx), e1 = __expf(a1 - mx);
            costRef -= isref ? (e0 / (e0 + e1)) : (e1 / (e0 + e1));
        }
        const int   base = ((t * NTILES + tile) * 16 + m) * 16;
        const float g    = ws[OFF_GSUM + t * N_ + n];
        const float numL = 2.0f * accL[base + n], psL = accL[base + N_];
        const float numS = 2.0f * accS[base + n], psS = accS[base + N_];
        diceL += (numL + 1.0f) / (psL + g + 1.0f);
        diceS += (numS + 1.0f) / (psS + g + 1.0f);
        const float iN = ws[OFF_IOUN + (t * NTILES + tile) * 16 + m];
        const float iD = ws[OFF_IOUD + (t * NTILES + tile) * 16 + m];
        iou += (iN + 1.0f) / (iD + 1.0f);
    }
    const float invT = 1.0f / (float)T_;
    out[tid] = (costRef - diceL - diceS - iou) * invT;
}

extern "C" void kernel_launch(void* const* d_in, const int* in_sizes, int n_in,
                              void* d_out, int out_size, void* d_ws, size_t ws_size,
                              hipStream_t stream) {
    (void)in_sizes; (void)n_in; (void)out_size; (void)ws_size;
    const float* predL = (const float*)d_in[0];   // [T,B,Q,72,72]
    const float* predS = (const float*)d_in[1];   // [T,B,Q,72,72]
    const float* refL  = (const float*)d_in[2];   // [T,B,Q,2]
    const float* refS  = (const float*)d_in[3];   // [T,B,Q,2]
    const float* tgt   = (const float*)d_in[4];   // [T,N,288,288]
    const int*   refI  = (const int*)d_in[5];     // [B]
    float* ws  = (float*)d_ws;
    float* out = (float*)d_out;                   // [B,Q,N] = [2,50,8]

    zero_ws_kernel<<<(WS_FLOATS + 255) / 256, 256, 0, stream>>>(ws, WS_FLOATS);
    gsum_kernel<<<T_ * N_, 256, 0, stream>>>(tgt, ws);
    fused_dice_wmma_kernel<<<dim3(T_ * NTILES, CSPLIT), 256, 0, stream>>>(
        predL, predS, tgt, ws);
    final_combine_kernel<<<(P_ * N_ + 255) / 256, 256, 0, stream>>>(
        refL, refS, refI, ws, out);
}
